// PairingLayer_84679575208565
// MI455X (gfx1250) — matrix operand order repaired and checked by reading, not compiled, not run on hardware
//
#include <hip/hip_runtime.h>

// Problem constants (from reference): B=32, N=1024, D=512, ONE_HOT=2*N-1
#define BB 32
#define NN 1024
#define DD 512
#define MT 32   // rows per block (2 WMMA row sub-tiles)

typedef __attribute__((ext_vector_type(16))) __bf16    v16bf;
typedef __attribute__((ext_vector_type(8)))  float     v8f;
typedef __attribute__((ext_vector_type(4)))  __bf16    v4bf;
typedef __attribute__((ext_vector_type(4)))  unsigned  u32x4;
typedef __attribute__((ext_vector_type(4)))  float     f32x4;

union FragBF { v16bf v; u32x4 q[2]; };

// ---------------------------------------------------------------------------
// Prep: si[b,n] = x·wi, sj[b,n] = x·wj, and bf16 copies xbf = x, abf = x*wij.
// One wave per row (32 lanes * 16 elems = 512). 8 waves / block.
// ---------------------------------------------------------------------------
__global__ __launch_bounds__(256) void pairing_prep_kernel(
    const float* __restrict__ x, const float* __restrict__ W,
    float* __restrict__ si, float* __restrict__ sj,
    __bf16* __restrict__ abf, __bf16* __restrict__ xbf) {
  const int lane = threadIdx.x & 31;
  const int wave = threadIdx.x >> 5;
  const int row  = blockIdx.x * 8 + wave;           // 0 .. B*N-1

  const float* __restrict__ xr  = x + (size_t)row * DD;
  const float* __restrict__ wi  = W;
  const float* __restrict__ wj  = W + DD;
  const float* __restrict__ wij = W + 2 * DD;

  float si_a = 0.f, sj_a = 0.f;
#pragma unroll
  for (int c = 0; c < 4; ++c) {
    const int idx = c * 128 + lane * 4;             // 128 consecutive elems/step
    f32x4 xv  = *(const f32x4*)(xr  + idx);
    f32x4 wiv = *(const f32x4*)(wi  + idx);
    f32x4 wjv = *(const f32x4*)(wj  + idx);
    f32x4 wv  = *(const f32x4*)(wij + idx);
    si_a += xv[0]*wiv[0] + xv[1]*wiv[1] + xv[2]*wiv[2] + xv[3]*wiv[3];
    sj_a += xv[0]*wjv[0] + xv[1]*wjv[1] + xv[2]*wjv[2] + xv[3]*wjv[3];
    v4bf xb, ab;
#pragma unroll
    for (int e = 0; e < 4; ++e) {
      xb[e] = (__bf16)xv[e];
      ab[e] = (__bf16)(xv[e] * wv[e]);
    }
    *(v4bf*)(xbf + (size_t)row * DD + idx) = xb;    // 8B store, 256B/wave contiguous
    *(v4bf*)(abf + (size_t)row * DD + idx) = ab;
  }
#pragma unroll
  for (int off = 16; off > 0; off >>= 1) {
    si_a += __shfl_xor(si_a, off, 32);
    sj_a += __shfl_xor(sj_a, off, 32);
  }
  if (lane == 0) { si[row] = si_a; sj[row] = sj_a; }
}

// ---------------------------------------------------------------------------
// Fused bf16-WMMA GEMM (scores) + bias/rel add + row softmax, write probs.
// Block: 256 thr (8 waves) = (batch b, 32-row tile). Wave w owns columns
// [w*128, w*128+128) as 8 col-tiles x 2 row-subtiles of 16x16 WMMA output;
// each B fragment feeds TWO WMMAs (halves L2 B-panel traffic vs MT=16).
// ---------------------------------------------------------------------------
__global__ __launch_bounds__(256) void pairing_score_softmax_kernel(
    const __bf16* __restrict__ abf, const __bf16* __restrict__ xbf,
    const float* __restrict__ si, const float* __restrict__ sj,
    const float* __restrict__ W, const float* __restrict__ bias,
    float* __restrict__ out) {
  __shared__ float sj_lds[NN];
  __shared__ float rel_lds[NN + MT];
  __shared__ float redmax[MT][8];
  __shared__ float redsum[MT][8];

  const int tid  = threadIdx.x;
  const int lane = tid & 31;
  const int wv   = tid >> 5;            // wave 0..7
  const int sel  = lane >> 4;           // half-wave select
  const int ln   = lane & 15;
  const int b    = blockIdx.y;
  const int r0   = blockIdx.x * MT;     // row tile base

  // Preload sj row-vector and the needed window of wpos (rel) into LDS.
  const float* __restrict__ wpos = W + 3 * DD;      // length 2*N-1
  for (int i = tid; i < NN; i += 256) sj_lds[i] = sj[b * NN + i];
  {
    const int base = (NN - MT) - r0;                // wpos idx of lds slot 0
    for (int i = tid; i < NN + MT - 1; i += 256) rel_lds[i] = wpos[base + i];
  }
  __syncthreads();

  const __bf16* __restrict__ Arow = abf + ((size_t)b * NN + r0) * DD;
  const __bf16* __restrict__ Brow = xbf + (size_t)b * NN * DD;
  const int colbase = wv * 128;

  v8f acc[2][8];
  {
    v8f z = {};
#pragma unroll
    for (int r = 0; r < 2; ++r)
#pragma unroll
      for (int t = 0; t < 8; ++t) acc[r][t] = z;
  }

  for (int k0 = 0; k0 < DD; k0 += 32) {
    // Two A fragments (16x32 bf16 each): lane = row ln (+16 per subtile),
    // halves split K per ISA layout (V0-3: K sel*8+0..7, V4-7: +16).
    FragBF af[2];
#pragma unroll
    for (int r = 0; r < 2; ++r) {
      const __bf16* p = Arow + (size_t)(r * 16 + ln) * DD + k0 + sel * 8;
      af[r].q[0] = *(const u32x4*)(p);
      af[r].q[1] = *(const u32x4*)(p + 16);
    }
#pragma unroll
    for (int t = 0; t < 8; ++t) {
      // B fragment (32x16 bf16): lane = column; x rows serve as B columns
      // (A @ x^T). Lane holds K = k0 + sel*8 + 0..7 and k0 + 16 + sel*8 + 0..7.
      FragBF bf;
      const __bf16* p = Brow + (size_t)(colbase + t * 16 + ln) * DD + k0 + sel * 8;
      bf.q[0] = *(const u32x4*)(p);
      bf.q[1] = *(const u32x4*)(p + 16);
      acc[0][t] = __builtin_amdgcn_wmma_f32_16x16x32_bf16(
          false, af[0].v, false, bf.v, (short)0, acc[0][t], false, false);
      acc[1][t] = __builtin_amdgcn_wmma_f32_16x16x32_bf16(
          false, af[1].v, false, bf.v, (short)0, acc[1][t], false, false);
    }
  }

  // --- epilogue: add si + sj + rel + bias, then row softmax over 1024 cols.
  // C/D layout: lane ln holds col = ln (+tile offset); VGPR v holds row
  // mm = r*16 + v + 8*sel within the 32-row block tile.
  const float b0 = bias[0];
  float sif[2][8];
#pragma unroll
  for (int r = 0; r < 2; ++r)
#pragma unroll
    for (int v = 0; v < 8; ++v)
      sif[r][v] = si[b * NN + r0 + r * 16 + v + 8 * sel] + b0;

  float rmax[2][8];
#pragma unroll
  for (int r = 0; r < 2; ++r)
#pragma unroll
    for (int v = 0; v < 8; ++v) rmax[r][v] = -3.402823466e38f;

#pragma unroll
  for (int t = 0; t < 8; ++t) {
    const int col = colbase + t * 16 + ln;
    const float sjc = sj_lds[col];
#pragma unroll
    for (int r = 0; r < 2; ++r)
#pragma unroll
      for (int v = 0; v < 8; ++v) {
        const int mm = r * 16 + v + 8 * sel;
        float s = acc[r][t][v] + sif[r][v] + sjc + rel_lds[col - mm + (MT - 1)];
        acc[r][t][v] = s;
        rmax[r][v] = fmaxf(rmax[r][v], s);
      }
  }
  // butterfly max across the 16 lanes of this half-wave (rows stay in-half)
#pragma unroll
  for (int off = 1; off < 16; off <<= 1)
#pragma unroll
    for (int r = 0; r < 2; ++r)
#pragma unroll
      for (int v = 0; v < 8; ++v)
        rmax[r][v] = fmaxf(rmax[r][v], __shfl_xor(rmax[r][v], off, 32));
  if (ln == 0)
#pragma unroll
    for (int r = 0; r < 2; ++r)
#pragma unroll
      for (int v = 0; v < 8; ++v) redmax[r * 16 + v + 8 * sel][wv] = rmax[r][v];
  __syncthreads();
#pragma unroll
  for (int r = 0; r < 2; ++r)
#pragma unroll
    for (int v = 0; v < 8; ++v) {
      const int mm = r * 16 + v + 8 * sel;
      float g = redmax[mm][0];
#pragma unroll
      for (int w2 = 1; w2 < 8; ++w2) g = fmaxf(g, redmax[mm][w2]);
      rmax[r][v] = g;
    }

  float rsum[2][8];
#pragma unroll
  for (int r = 0; r < 2; ++r)
#pragma unroll
    for (int v = 0; v < 8; ++v) rsum[r][v] = 0.f;
#pragma unroll
  for (int t = 0; t < 8; ++t)
#pragma unroll
    for (int r = 0; r < 2; ++r)
#pragma unroll
      for (int v = 0; v < 8; ++v) {
        float e = __expf(acc[r][t][v] - rmax[r][v]);
        acc[r][t][v] = e;
        rsum[r][v] += e;
      }
#pragma unroll
  for (int off = 1; off < 16; off <<= 1)
#pragma unroll
    for (int r = 0; r < 2; ++r)
#pragma unroll
      for (int v = 0; v < 8; ++v) rsum[r][v] += __shfl_xor(rsum[r][v], off, 32);
  if (ln == 0)
#pragma unroll
    for (int r = 0; r < 2; ++r)
#pragma unroll
      for (int v = 0; v < 8; ++v) redsum[r * 16 + v + 8 * sel][wv] = rsum[r][v];
  __syncthreads();
#pragma unroll
  for (int r = 0; r < 2; ++r)
#pragma unroll
    for (int v = 0; v < 8; ++v) {
      const int mm = r * 16 + v + 8 * sel;
      float g = 0.f;
#pragma unroll
      for (int w2 = 0; w2 < 8; ++w2) g += redsum[mm][w2];
      rsum[r][v] = 1.0f / g;
    }

  float* __restrict__ orow = out + ((size_t)b * NN + r0) * NN;
#pragma unroll
  for (int t = 0; t < 8; ++t) {
    const int col = colbase + t * 16 + ln;
#pragma unroll
    for (int r = 0; r < 2; ++r)
#pragma unroll
      for (int v = 0; v < 8; ++v) {
        const int mm = r * 16 + v + 8 * sel;
        orow[(size_t)mm * NN + col] = acc[r][t][v] * rsum[r][v];
      }
  }
}

// ---------------------------------------------------------------------------
extern "C" void kernel_launch(void* const* d_in, const int* in_sizes, int n_in,
                              void* d_out, int out_size, void* d_ws, size_t ws_size,
                              hipStream_t stream) {
  (void)in_sizes; (void)n_in; (void)out_size; (void)ws_size;
  const float* x = (const float*)d_in[0];   // [B,N,D] f32
  const float* W = (const float*)d_in[1];   // [3*D + 2N-1] f32
  const float* b = (const float*)d_in[2];   // [1] f32
  float* out = (float*)d_out;               // [B,N,N] f32

  // workspace layout: si | sj | abf(bf16) | xbf(bf16)  (~64.3 MB)
  char* ws = (char*)d_ws;
  float*  si  = (float*)ws;                                   // B*N
  float*  sj  = si + (size_t)BB * NN;                         // B*N
  __bf16* abf = (__bf16*)(ws + 2u * BB * NN * sizeof(float)); // B*N*D
  __bf16* xbf = abf + (size_t)BB * NN * DD;                   // B*N*D

  pairing_prep_kernel<<<(BB * NN) / 8, 256, 0, stream>>>(x, W, si, sj, abf, xbf);
  pairing_score_softmax_kernel<<<dim3(NN / MT, BB), 256, 0, stream>>>(
      abf, xbf, si, sj, W, b, out);
}
